// MultiHeadAttn_23244363006590
// MI455X (gfx1250) — matrix operand order repaired and checked by reading, not compiled, not run on hardware
//
#include <hip/hip_runtime.h>
#include <hip/hip_bf16.h>

// ---------------------------------------------------------------------------
// MI455X (gfx1250) fused multi-head neighbor-attention block.
//  - Heavy GEMMs on v_wmma_f32_16x16x32_bf16 (wave32, f32 accumulate)
//  - LayerNorm fused into the A-fragment load (per-row stats precomputed)
//  - logit = row.W2 dot fused into the GEMM epilogue (shuffle + atomicAdd)
//  - B panel double-buffered in LDS via gfx1250 async global->LDS DMA
//    (global_load_async_to_lds_b128 + s_wait_asynccnt), consumed by ds loads
//  - Block tile 32(M) x 768(N, full width) so y is streamed exactly once/head
// ---------------------------------------------------------------------------

typedef __attribute__((ext_vector_type(16))) __bf16 v16bf;
typedef __attribute__((ext_vector_type(8)))  __bf16 bf16x8;
typedef __attribute__((ext_vector_type(8)))  float  v8f;

#define E_DIM   768
#define H_NUM   4
#define BS_TOK  2048
#define K_NBR   36
#define MROWS_Y (BS_TOK * K_NBR)   // 73728
#define CAT_W   (H_NUM * E_DIM)    // 3072
#define BK      32                 // K chunk staged per LDS buffer
#define NCOL    768                // full N width staged per chunk
#define NT      12                 // WMMA N-tiles per wave (192 cols)

__device__ __forceinline__ float gelu_tanh(float v) {
    const float c0 = 0.7978845608028654f;   // sqrt(2/pi)
    const float c1 = 0.044715f;
    float u = c0 * (v + c1 * v * v * v);
    return 0.5f * v * (1.0f + tanhf(u));
}

// gfx1250 async global->LDS DMA: 16 bytes per active lane, lane-private LDS
// destination offset.  Tracked by ASYNCcnt (ISA 10: GLOBAL_LOAD_ASYNC_TO_LDS).
__device__ __forceinline__ void async_g2l_b128(unsigned lds_off, const void* gptr) {
    asm volatile("global_load_async_to_lds_b128 %0, %1, off"
                 :: "v"(lds_off), "v"(gptr) : "memory");
}
__device__ __forceinline__ void wait_asynccnt0() {
    asm volatile("s_wait_asynccnt 0x0" ::: "memory");
}

// Stage B chunk: Bt[col][k0..k0+31] for col in 0..767 into sbuf ([col][32] bf16,
// 64 B/column).  3072 x 16B transfers -> 12 per thread (256 threads).
__device__ __forceinline__ void stage_b_chunk(const __bf16* __restrict__ Bt, int Ktot,
                                              int k0, __bf16* sbuf, int tid) {
    unsigned base = (unsigned)(size_t)sbuf;     // LDS byte offset (addr[31:0])
    #pragma unroll
    for (int j = 0; j < 12; ++j) {
        int e   = j * 256 + tid;                // 0..3071
        int col = e >> 2;
        int seg = e & 3;                        // 4 x 16B per column
        const __bf16* g = Bt + (size_t)col * Ktot + k0 + seg * 8;
        unsigned lds = base + (unsigned)(col * 64 + seg * 16);
        async_g2l_b128(lds, g);
    }
}

// ---------------------------------------------------------------------------
// Per-row LayerNorm statistics (mean, rstd) for y rows and x rows.
// ---------------------------------------------------------------------------
__global__ __launch_bounds__(256)
void row_stats_kernel(const float* __restrict__ y, const float* __restrict__ x,
                      float* __restrict__ muY, float* __restrict__ rsY,
                      float* __restrict__ muX, float* __restrict__ rsX) {
    int wave = (int)((blockIdx.x * blockDim.x + threadIdx.x) >> 5);
    int lane = threadIdx.x & 31;
    const int totalRows = MROWS_Y + BS_TOK;
    if (wave >= totalRows) return;
    const float* row = (wave < MROWS_Y) ? (y + (size_t)wave * E_DIM)
                                        : (x + (size_t)(wave - MROWS_Y) * E_DIM);
    float s = 0.f, ss = 0.f;
    for (int j = lane; j < E_DIM; j += 32) {
        float v = row[j];
        s += v; ss += v * v;
    }
    #pragma unroll
    for (int m = 16; m >= 1; m >>= 1) {
        s  += __shfl_xor(s,  m, 32);
        ss += __shfl_xor(ss, m, 32);
    }
    float mu  = s * (1.0f / E_DIM);
    float var = ss * (1.0f / E_DIM) - mu * mu;
    float rstd = rsqrtf(var + 1e-5f);
    if (lane == 0) {
        if (wave < MROWS_Y) { muY[wave] = mu; rsY[wave] = rstd; }
        else                { muX[wave - MROWS_Y] = mu; rsX[wave - MROWS_Y] = rstd; }
    }
}

// ---------------------------------------------------------------------------
// Transpose + f32->bf16: in [K,N] row-major f32 -> out [N,K] bf16.
// ---------------------------------------------------------------------------
__global__ void transpose_bf16_kernel(const float* __restrict__ in,
                                      __bf16* __restrict__ out, int K, int N) {
    size_t total = (size_t)K * N;
    for (size_t i = blockIdx.x * (size_t)blockDim.x + threadIdx.x; i < total;
         i += (size_t)gridDim.x * blockDim.x) {
        size_t n = i / K, k = i % K;
        out[i] = (__bf16)in[k * (size_t)N + n];
    }
}

// ---------------------------------------------------------------------------
// Fused LN -> GEMM(bf16 WMMA) -> (+b1) -> bf16 store + fused logit dot.
//   A  : f32 [M, 768] (LN applied on the fly);  Bt : bf16 [768(N)][768(K)]
// Block: 32 rows x 768 cols, 8 waves as 2(M) x 4(N); wave strip 16 x 192.
// B chunks (768x32) double-buffered in LDS via async DMA.
// ---------------------------------------------------------------------------
__global__ __launch_bounds__(256)
void gemm_ln_attn_kernel(const float* __restrict__ A,
                         const float* __restrict__ mu, const float* __restrict__ rs,
                         const float* __restrict__ gamma, const float* __restrict__ beta,
                         const __bf16* __restrict__ Bt,
                         const float* __restrict__ b1,
                         const float* __restrict__ w2,
                         __bf16* __restrict__ out,    // [M, 768] bf16
                         float* __restrict__ logit) { // [M] f32, pre-zeroed
    __shared__ __bf16 sBuf[2][NCOL * BK];             // 2 x 48 KB
    __shared__ float  sGm[E_DIM], sBe[E_DIM];         // 6 KB

    const int tid = threadIdx.x;
    for (int i = tid; i < E_DIM; i += 256) { sGm[i] = gamma[i]; sBe[i] = beta[i]; }

    const int wave  = tid >> 5;
    const int lane  = tid & 31;
    const int ln15  = lane & 15;
    const int half  = lane >> 4;                // 0 or 1
    const int waveM = wave >> 2;                // 0..1
    const int waveN = wave & 3;                 // 0..3
    const int M0 = blockIdx.x * 32 + waveM * 16;
    const int N0 = waveN * (NT * 16);           // 192-col strip

    const int rowA = M0 + ln15;
    const float rmu = mu[rowA];
    const float rrs = rs[rowA];
    const float* arow = A + (size_t)rowA * E_DIM;
    const int ha = half * 16;                   // A K-offset of this lane half

    v8f acc[NT];
    #pragma unroll
    for (int t = 0; t < NT; ++t) acc[t] = (v8f){};

    const int NC = E_DIM / BK;                  // 24 chunks
    stage_b_chunk(Bt, E_DIM, 0, sBuf[0], tid);  // prologue

    for (int c = 0; c < NC; ++c) {
        wait_asynccnt0();
        __syncthreads();                        // chunk c visible to all waves
        if (c + 1 < NC)
            stage_b_chunk(Bt, E_DIM, (c + 1) * BK, sBuf[(c + 1) & 1], tid);

        // ---- A fragment: f32 load + LN + bf16 convert ----
        const int k0 = c * BK;
        const int ka = k0 + ha;
        v8f y0 = *(const v8f*)(arow + ka);
        v8f y1 = *(const v8f*)(arow + ka + 8);
        __builtin_prefetch(arow + ka + BK, 0, 3);
        v16bf af;
        #pragma unroll
        for (int j = 0; j < 8; ++j) {
            float v0 = (y0[j] - rmu) * rrs * sGm[ka + j]      + sBe[ka + j];
            float v1 = (y1[j] - rmu) * rrs * sGm[ka + 8 + j] + sBe[ka + 8 + j];
            af[j]     = (__bf16)v0;
            af[j + 8] = (__bf16)v1;
        }
        // ---- 12 WMMAs from the LDS-staged B chunk ----
        const __bf16* sb = sBuf[c & 1] + (size_t)half * 16;
        #pragma unroll
        for (int t = 0; t < NT; ++t) {
            v16bf bfr = *(const v16bf*)(sb + (size_t)(N0 + t * 16 + ln15) * BK);
            acc[t] = __builtin_amdgcn_wmma_f32_16x16x32_bf16(
                false, af, false, bfr, (short)0, acc[t], false, false);
        }
        __syncthreads();                        // done reading buf before reuse
    }

    // ---- Epilogue: +b1, bf16 store, fused logit partial dot ----
    float pl[8];
    #pragma unroll
    for (int r = 0; r < 8; ++r) pl[r] = 0.f;
    const int rowC = M0 + half * 8;             // C layout: VGPR r -> M=r (+8 hi half)

    #pragma unroll
    for (int t = 0; t < NT; ++t) {
        int col = N0 + t * 16 + ln15;
        float b1v = b1[col];
        float w2v = w2[col];
        #pragma unroll
        for (int r = 0; r < 8; ++r) {
            float val = acc[t][r] + b1v;
            out[(size_t)(rowC + r) * E_DIM + col] = (__bf16)val;
            pl[r] += val * w2v;
        }
    }
    #pragma unroll
    for (int r = 0; r < 8; ++r) {
        float t = pl[r];
        #pragma unroll
        for (int m = 1; m < 16; m <<= 1) t += __shfl_xor(t, m, 32);
        if (ln15 == 0) atomicAdd(&logit[rowC + r], t);
    }
}

// ---------------------------------------------------------------------------
// Per-token softmax over 36 neighbors + weighted sum of bf16 y_ rows,
// then GELU(x + a) into the bf16 concat buffer at the head's column offset.
// ---------------------------------------------------------------------------
__global__ __launch_bounds__(256)
void attn_softmax_kernel(const float* __restrict__ xf,
                         const __bf16* __restrict__ ybf,    // [BS*K, E]
                         const float* __restrict__ logitx,  // [BS]
                         const float* __restrict__ logity,  // [BS*K]
                         const float* __restrict__ b2, int h,
                         __bf16* __restrict__ cat) {        // [BS, CAT_W]
    __shared__ float sl[K_NBR];
    __shared__ float p[K_NBR];
    const int n = blockIdx.x;
    const int tid = threadIdx.x;
    if (tid < K_NBR) sl[tid] = logitx[n] + logity[n * K_NBR + tid] + b2[h];
    __syncthreads();
    if (tid == 0) {
        float mx = -1e30f;
        for (int k = 0; k < K_NBR; ++k) mx = fmaxf(mx, sl[k]);
        float s = 0.f;
        for (int k = 0; k < K_NBR; ++k) { float e = __expf(sl[k] - mx); p[k] = e; s += e; }
        float inv = 1.f / s;
        for (int k = 0; k < K_NBR; ++k) p[k] *= inv;
    }
    __syncthreads();
    const int headOff = h * E_DIM;
    for (int e = tid; e < E_DIM; e += 256) {
        float a = 0.f;
        const __bf16* yb = ybf + (size_t)n * K_NBR * E_DIM + e;
        #pragma unroll 4
        for (int k = 0; k < K_NBR; ++k) a += p[k] * (float)yb[(size_t)k * E_DIM];
        float o = gelu_tanh(xf[(size_t)n * E_DIM + e] + a);
        cat[(size_t)n * CAT_W + headOff + e] = (__bf16)o;
    }
}

// ---------------------------------------------------------------------------
// Final GEMM: out = x + gelu(cat @ Wm + bm).
//   A : bf16 [2048, 3072];  Bt : bf16 [768][3072].  Same async-staged tiling.
// ---------------------------------------------------------------------------
__global__ __launch_bounds__(256)
void gemm_final_kernel(const __bf16* __restrict__ Abf,
                       const __bf16* __restrict__ Bt,
                       const float* __restrict__ bm,
                       const float* __restrict__ x,
                       float* __restrict__ out) {
    __shared__ __bf16 sBuf[2][NCOL * BK];

    const int tid   = threadIdx.x;
    const int wave  = tid >> 5;
    const int lane  = tid & 31;
    const int ln15  = lane & 15;
    const int half  = lane >> 4;
    const int waveM = wave >> 2;
    const int waveN = wave & 3;
    const int M0 = blockIdx.x * 32 + waveM * 16;
    const int N0 = waveN * (NT * 16);

    const __bf16* arow = Abf + (size_t)(M0 + ln15) * CAT_W;
    const int ha = half * 16;

    v8f acc[NT];
    #pragma unroll
    for (int t = 0; t < NT; ++t) acc[t] = (v8f){};

    const int NC = CAT_W / BK;                  // 96 chunks
    stage_b_chunk(Bt, CAT_W, 0, sBuf[0], tid);

    for (int c = 0; c < NC; ++c) {
        wait_asynccnt0();
        __syncthreads();
        if (c + 1 < NC)
            stage_b_chunk(Bt, CAT_W, (c + 1) * BK, sBuf[(c + 1) & 1], tid);

        const int ka = c * BK + ha;
        bf16x8 lo = *(const bf16x8*)(arow + ka);
        bf16x8 hi = *(const bf16x8*)(arow + ka + 8);
        __builtin_prefetch(arow + ka + BK, 0, 3);
        v16bf af = __builtin_shufflevector(lo, hi, 0, 1, 2, 3, 4, 5, 6, 7,
                                                   8, 9, 10, 11, 12, 13, 14, 15);
        const __bf16* sb = sBuf[c & 1] + (size_t)half * 16;
        #pragma unroll
        for (int t = 0; t < NT; ++t) {
            v16bf bfr = *(const v16bf*)(sb + (size_t)(N0 + t * 16 + ln15) * BK);
            acc[t] = __builtin_amdgcn_wmma_f32_16x16x32_bf16(
                false, af, false, bfr, (short)0, acc[t], false, false);
        }
        __syncthreads();
    }

    const int rowC = M0 + half * 8;
    #pragma unroll
    for (int t = 0; t < NT; ++t) {
        int col = N0 + t * 16 + ln15;
        float bmv = bm[col];
        #pragma unroll
        for (int r = 0; r < 8; ++r) {
            size_t idx = (size_t)(rowC + r) * E_DIM + col;
            out[idx] = x[idx] + gelu_tanh(acc[t][r] + bmv);
        }
    }
}

// ---------------------------------------------------------------------------
extern "C" void kernel_launch(void* const* d_in, const int* in_sizes, int n_in,
                              void* d_out, int out_size, void* d_ws, size_t ws_size,
                              hipStream_t stream) {
    (void)in_sizes; (void)n_in; (void)out_size; (void)ws_size;
    const float* x     = (const float*)d_in[0];   // [32,64,768]
    const float* y     = (const float*)d_in[1];   // [2048,36,768]
    const float* gamma = (const float*)d_in[2];   // [4,768]
    const float* beta  = (const float*)d_in[3];   // [4,768]
    const float* W1    = (const float*)d_in[4];   // [4,768,768]
    const float* b1    = (const float*)d_in[5];   // [4,768]
    const float* W2x   = (const float*)d_in[6];   // [4,768]
    const float* W2y   = (const float*)d_in[7];   // [4,768]
    const float* b2    = (const float*)d_in[8];   // [4]
    const float* Wm    = (const float*)d_in[9];   // [3072,768]
    const float* bm    = (const float*)d_in[10];  // [768]
    float* out = (float*)d_out;

    // Workspace carve-up (~136 MB), 256B-aligned chunks.
    char* w = (char*)d_ws;
    auto carve = [&](size_t bytes) -> char* {
        char* p = w; w += (bytes + 255) & ~(size_t)255; return p;
    };
    float*  muY    = (float*)carve((size_t)MROWS_Y * 4);
    float*  rsY    = (float*)carve((size_t)MROWS_Y * 4);
    float*  muX    = (float*)carve((size_t)BS_TOK * 4);
    float*  rsX    = (float*)carve((size_t)BS_TOK * 4);
    float*  logity = (float*)carve((size_t)MROWS_Y * 4);
    float*  logitx = (float*)carve((size_t)BS_TOK * 4);
    __bf16* W1t    = (__bf16*)carve((size_t)E_DIM * E_DIM * 2);
    __bf16* Wmt    = (__bf16*)carve((size_t)CAT_W * E_DIM * 2);
    __bf16* xbf    = (__bf16*)carve((size_t)BS_TOK * E_DIM * 2);
    __bf16* cat    = (__bf16*)carve((size_t)BS_TOK * CAT_W * 2);
    __bf16* ybf    = (__bf16*)carve((size_t)MROWS_Y * E_DIM * 2);

    {   // LN row statistics for y and x (once).
        int waves = MROWS_Y + BS_TOK;
        int blocks = (waves + 7) / 8;
        row_stats_kernel<<<blocks, 256, 0, stream>>>(y, x, muY, rsY, muX, rsX);
    }
    // Wm -> transposed bf16 (once).
    transpose_bf16_kernel<<<1024, 256, 0, stream>>>(Wm, Wmt, CAT_W, E_DIM);

    for (int h = 0; h < H_NUM; ++h) {
        transpose_bf16_kernel<<<1024, 256, 0, stream>>>(W1 + (size_t)h * E_DIM * E_DIM,
                                                        W1t, E_DIM, E_DIM);
        hipMemsetAsync(logity, 0, (size_t)MROWS_Y * 4, stream);
        hipMemsetAsync(logitx, 0, (size_t)BS_TOK * 4, stream);

        gemm_ln_attn_kernel<<<MROWS_Y / 32, 256, 0, stream>>>(
            y, muY, rsY, gamma + h * E_DIM, beta + h * E_DIM,
            W1t, b1 + h * E_DIM, W2y + h * E_DIM, ybf, logity);
        gemm_ln_attn_kernel<<<BS_TOK / 32, 256, 0, stream>>>(
            x, muX, rsX, gamma + h * E_DIM, beta + h * E_DIM,
            W1t, b1 + h * E_DIM, W2x + h * E_DIM, xbf, logitx);
        attn_softmax_kernel<<<BS_TOK, 256, 0, stream>>>(x, ybf, logitx, logity, b2, h, cat);
    }

    gemm_final_kernel<<<BS_TOK / 32, 256, 0, stream>>>(cat, Wmt, bm, x, out);
}
